// GradSolver_58102317580919
// MI455X (gfx1250) — matrix operand order repaired and checked by reading, not compiled, not run on hardware
//
#include <hip/hip_runtime.h>

#define BB   16
#define NPG  2048
#define FF   1024
#define EPG  65536
#define NN   (BB * NPG)      // 32768
#define EE   (BB * EPG)      // 1048576
#define STEPS 64
#define PELEMS ((size_t)BB * NPG * FF)   // 33,554,432

typedef __attribute__((ext_vector_type(16))) __bf16 v16bf;
typedef __attribute__((ext_vector_type(8)))  float  v8f;

union V16 { v16bf v; __bf16 e[16]; };

// 16-bit A-fragment (16xK, lane = row M): ISA 7.12.2
//   lanes 0-15 : halves 0..7 -> K=0..7,  halves 8..15 -> K=16..23
//   lanes 16-31: halves 0..7 -> K=8..15, halves 8..15 -> K=24..31
__device__ __forceinline__ int amap(int j, int hi) {
  return (j & 7) + ((j >= 8) ? 16 : 0) + hi * 8;
}
// contiguous K order (this code's choice, used for A and B in k_coef_t): consecutive K per lane-half
//   lanes 0-15: halves -> K=0..15 ; lanes 16-31: halves -> K=16..31
__device__ __forceinline__ int bmap(int j, int hi) {
  return j + hi * 16;
}

__device__ __forceinline__ __bf16 ldbf(const float*  p, size_t i) { return (__bf16)p[i]; }
__device__ __forceinline__ __bf16 ldbf(const __bf16* p, size_t i) { return p[i]; }

// -------------------------------------------- one-time proj f32 -> bf16 staging
__global__ void k_cvt(const float* __restrict__ P, __bf16* __restrict__ P16) {
  size_t i = ((size_t)blockIdx.x * 256 + threadIdx.x) * 8;
  float4 a = *(const float4*)(P + i);
  float4 b = *(const float4*)(P + i + 4);
  union { __bf16 h[8]; uint4 u; } pk;
  pk.h[0] = (__bf16)a.x; pk.h[1] = (__bf16)a.y;
  pk.h[2] = (__bf16)a.z; pk.h[3] = (__bf16)a.w;
  pk.h[4] = (__bf16)b.x; pk.h[5] = (__bf16)b.y;
  pk.h[6] = (__bf16)b.z; pk.h[7] = (__bf16)b.w;
  *(uint4*)(void*)(P16 + i) = pk.u;
}

// ---------------------------------------------------------------- init
__global__ void k_init(const float* __restrict__ x_start,
                       float* __restrict__ x, float* __restrict__ best_x,
                       float* __restrict__ flag) {
  int i = blockIdx.x * 256 + threadIdx.x;
  if (i < NN) {
    float v = x_start[i];
    x[i] = v;
    best_x[i] = v;
  }
  if (i < BB) flag[i] = 0.0f;
}

// ------------------------------------------------- direction (+ flush best_x)
__global__ void k_dir(const float* __restrict__ x, const float* __restrict__ rhs,
                      const float* __restrict__ flag, float* __restrict__ best_x,
                      float* __restrict__ d, float tau) {
  int i = blockIdx.x * 256 + threadIdx.x;
  if (i >= NN) return;
  int b = i >> 11;
  float xv = x[i];
  if (flag[b] != 0.0f) best_x[i] = xv;
  d[i] = -xv + rhs[i] + 0.1f * tau / (xv + tau);
}

// --------------------------------------------- coef[b,f] = sum_n P[b,n,f]*d[b,n]
// grid (FF/16, BB), 256 threads = 8 waves; each wave reduces a 256-wide n slice.
// A = d replicated over all 16 rows -> C[m][c] = coef(f0+c) for every m.
template <typename T>
__global__ void k_coef_t(const T* __restrict__ P, const float* __restrict__ d,
                         float* __restrict__ coef) {
  int f0 = blockIdx.x * 16;
  int b  = blockIdx.y;
  int lane = threadIdx.x & 31;
  int w    = threadIdx.x >> 5;
  int hi   = lane >> 4;
  int col  = lane & 15;
  const T*     Pb = P + (size_t)b * NPG * FF;
  const float* db = d + b * NPG;

  v8f acc = {};
  for (int it = 0; it < 8; ++it) {
    int n0 = w * 256 + it * 32;
    // prefetch one K-chunk ahead (speculative; dropped if OOB)
    __builtin_prefetch(Pb + (size_t)(n0 + 32 + hi * 16) * FF + f0 + col, 0, 0);
    V16 a, bm;
#pragma unroll
    for (int j = 0; j < 16; ++j)
      a.e[j] = (__bf16)db[n0 + bmap(j, hi)];                 // replicated vector
#pragma unroll
    for (int j = 0; j < 16; ++j)
      bm.e[j] = ldbf(Pb, (size_t)(n0 + bmap(j, hi)) * FF + f0 + col); // P tile (Kx16)
    acc = __builtin_amdgcn_wmma_f32_16x16x32_bf16(false, a.v, false, bm.v,
                                                  (short)0, acc, false, false);
  }

  __shared__ float red[8 * 16];
  if (lane < 16) red[w * 16 + lane] = acc[0];   // C[0][lane] = coef(f0+lane) partial
  __syncthreads();
  if (threadIdx.x < 16) {
    float s = 0.0f;
    for (int w2 = 0; w2 < 8; ++w2) s += red[w2 * 16 + threadIdx.x];
    coef[b * FF + f0 + threadIdx.x] = s;
  }
}

// ---------------------------------- pred[b,n] = sum_f P[b,n,f]*coef[b,f] + ratio min
// grid (NPG/16, BB), 256 threads = 8 waves; each wave reduces a 128-wide f slice.
// A = P tile (16 rows of n), B = coef replicated over all 16 columns.
template <typename T>
__global__ void k_pred_t(const T* __restrict__ P, const float* __restrict__ coef,
                         const float* __restrict__ x, float* __restrict__ pred,
                         float* __restrict__ minpart) {
  int t0 = blockIdx.x;            // n tile
  int b  = blockIdx.y;
  int n0 = t0 * 16;
  int lane = threadIdx.x & 31;
  int w    = threadIdx.x >> 5;
  int hi   = lane >> 4;
  int m    = lane & 15;
  const T*     Pb = P + (size_t)b * NPG * FF;
  const float* cb = coef + b * FF;
  size_t rowbase = (size_t)(n0 + m) * FF;

  v8f acc = {};
  for (int it = 0; it < 4; ++it) {
    int f0 = w * 128 + it * 32;
    __builtin_prefetch(Pb + rowbase + f0 + 32, 0, 0);
    V16 a, bm;
    // A-fragment row loads: two contiguous 8-half groups per lane (K map)
#pragma unroll
    for (int k = 0; k < 8; ++k)
      a.e[k] = ldbf(Pb, rowbase + f0 + hi * 8 + k);          // K = hi*8 + 0..7
#pragma unroll
    for (int k = 0; k < 8; ++k)
      a.e[8 + k] = ldbf(Pb, rowbase + f0 + 16 + hi * 8 + k); // K = 16 + hi*8 + 0..7
#pragma unroll
    for (int j = 0; j < 16; ++j)
      bm.e[j] = (__bf16)cb[f0 + amap(j, hi)];                // replicated coef
    acc = __builtin_amdgcn_wmma_f32_16x16x32_bf16(false, a.v, false, bm.v,
                                                  (short)0, acc, false, false);
  }

  __shared__ float red[8 * 16];
  __shared__ float red2[16];
  // lane 0 holds pred[n0+0..7] in acc[0..7]; lane 16 holds pred[n0+8..15]
  if (m == 0) {
#pragma unroll
    for (int v = 0; v < 8; ++v) red[w * 16 + hi * 8 + v] = acc[v];
  }
  __syncthreads();
  if (threadIdx.x < 16) {
    float s = 0.0f;
    for (int w2 = 0; w2 < 8; ++w2) s += red[w2 * 16 + threadIdx.x];
    int n = b * NPG + n0 + threadIdx.x;
    pred[n] = s;
    red2[threadIdx.x] = (s < 0.0f) ? (x[n] / (-s)) : 1.0f;   // STEP_ALPHA = 1.0
  }
  __syncthreads();
  if (threadIdx.x == 0) {
    float mn = red2[0];
    for (int i = 1; i < 16; ++i) mn = fminf(mn, red2[i]);
    minpart[b * (NPG / 16) + t0] = mn;
  }
}

// ------------------------------------------------------------- alpha per batch
__global__ void k_alpha(const float* __restrict__ minpart, float* __restrict__ alpha) {
  int b = blockIdx.x;
  int t = threadIdx.x;            // 128 threads
  __shared__ float sm[128];
  sm[t] = minpart[b * 128 + t];
  __syncthreads();
  for (int st = 64; st > 0; st >>= 1) {
    if (t < st) sm[t] = fminf(sm[t], sm[t + st]);
    __syncthreads();
  }
  if (t == 0) alpha[b] = fminf(sm[0], 1.0f) * 0.995f;
}

// ------------------------------------------- x update + linear-term partials
__global__ void k_xupd(float* __restrict__ x, const float* __restrict__ pred,
                       const float* __restrict__ alpha, const float* __restrict__ q,
                       float* __restrict__ linpart) {
  int i = blockIdx.x * 256 + threadIdx.x;   // 128 blocks cover NN
  int b = i >> 11;
  float xn = x[i] + alpha[b] * pred[i];
  x[i] = xn;
  __shared__ float sm[256];
  sm[threadIdx.x] = q[i] * xn;
  __syncthreads();
  for (int st = 128; st > 0; st >>= 1) {
    if (threadIdx.x < st) sm[threadIdx.x] += sm[threadIdx.x + st];
    __syncthreads();
  }
  if (threadIdx.x == 0) linpart[blockIdx.x] = sm[0];   // blockIdx.x = b*8 + j
}

// ------------------------------------------- linear partials only (init pass)
__global__ void k_lin_init(const float* __restrict__ q, const float* __restrict__ x,
                           float* __restrict__ linpart) {
  int i = blockIdx.x * 256 + threadIdx.x;
  __shared__ float sm[256];
  sm[threadIdx.x] = q[i] * x[i];
  __syncthreads();
  for (int st = 128; st > 0; st >>= 1) {
    if (threadIdx.x < st) sm[threadIdx.x] += sm[threadIdx.x + st];
    __syncthreads();
  }
  if (threadIdx.x == 0) linpart[blockIdx.x] = sm[0];
}

// ------------------------------------------------------ quadratic edge partials
__global__ void k_quad(const int* __restrict__ src, const int* __restrict__ dst,
                       const float* __restrict__ ew, const float* __restrict__ x,
                       float* __restrict__ quadpart) {
  int b   = blockIdx.y;
  int blk = blockIdx.x;                   // 64 blocks, 1024 edges each
  int e0  = b * EPG + blk * 1024;
  float s = 0.0f;
  for (int t = threadIdx.x; t < 1024; t += 256) {
    int e = e0 + t;
    s += ew[e] * x[src[e]] * x[dst[e]];
  }
  __shared__ float sm[256];
  sm[threadIdx.x] = s;
  __syncthreads();
  for (int st = 128; st > 0; st >>= 1) {
    if (threadIdx.x < st) sm[threadIdx.x] += sm[threadIdx.x + st];
    __syncthreads();
  }
  if (threadIdx.x == 0) quadpart[b * 64 + blk] = sm[0];
}

// ---------------------------------------------- objective finalize / best track
__global__ void k_objfin(const float* __restrict__ quadpart,
                         const float* __restrict__ linpart,
                         float* __restrict__ best_obj, float* __restrict__ flag,
                         const float* __restrict__ obj_solution,
                         float* __restrict__ out_best, float* __restrict__ out_gaps,
                         int s) {
  int b = threadIdx.x;
  if (b >= BB) return;
  float quad = 0.0f;
  for (int i = 0; i < 64; ++i) quad += quadpart[b * 64 + i];
  float lin = 0.0f;
  for (int j = 0; j < 8; ++j) lin += linpart[b * 8 + j];
  float obj = 0.5f * quad + lin;
  if (s < 0) {                 // init: best_obj0 = qp_obj(x_start)
    best_obj[b] = obj;
    flag[b] = 0.0f;
    return;
  }
  float bo = best_obj[b];
  bool better = obj < bo;
  bo = better ? obj : bo;
  best_obj[b] = bo;
  flag[b] = better ? 1.0f : 0.0f;
  float opt = obj_solution[b];
  float gap = fabsf((opt - bo) / opt);
  out_gaps[b * STEPS + s] = gap;
  if (s == STEPS - 1) out_best[b] = gap;
}

// ----------------------------------------------------- final best_x flush
__global__ void k_flush(const float* __restrict__ x, const float* __restrict__ flag,
                        float* __restrict__ best_x) {
  int i = blockIdx.x * 256 + threadIdx.x;
  if (i >= NN) return;
  if (flag[i >> 11] != 0.0f) best_x[i] = x[i];
}

extern "C" void kernel_launch(void* const* d_in, const int* in_sizes, int n_in,
                              void* d_out, int out_size, void* d_ws, size_t ws_size,
                              hipStream_t stream) {
  (void)in_sizes; (void)n_in; (void)out_size;
  const float* x_start = (const float*)d_in[0];
  const float* rhs     = (const float*)d_in[1];
  const float* proj    = (const float*)d_in[2];
  const float* q       = (const float*)d_in[3];
  const float* obj_sol = (const float*)d_in[4];
  const int*   eidx    = (const int*)d_in[5];   // [2, E] flattened
  const float* ew      = (const float*)d_in[6];
  // d_in[7] (vals_batch) not needed: batch = node >> 11

  float* out      = (float*)d_out;
  float* best_x   = out;                 // NN
  float* out_best = out + NN;            // BB
  float* out_gaps = out + NN + BB;       // BB*STEPS

  float* ws       = (float*)d_ws;
  float* x        = ws;            ws += NN;
  float* d        = ws;            ws += NN;
  float* pred     = ws;            ws += NN;
  float* coef     = ws;            ws += BB * FF;
  float* quadpart = ws;            ws += BB * 64;
  float* linpart  = ws;            ws += 128;
  float* minpart  = ws;            ws += BB * (NPG / 16);
  float* alpha    = ws;            ws += BB;
  float* best_obj = ws;            ws += BB;
  float* flag     = ws;            ws += BB;
  // pad to 16B boundary for the bf16 proj copy
  size_t small_floats = (size_t)(ws - (float*)d_ws);
  small_floats = (small_floats + 3) & ~(size_t)3;
  __bf16* proj16 = (__bf16*)((float*)d_ws + small_floats);
  size_t need = small_floats * 4 + PELEMS * 2;
  bool use16 = (ws_size >= need);   // deterministic: sizes fixed per harness

  const int* src = eidx;
  const int* dst = eidx + EE;

  if (use16)   // one-time bf16 staging of proj (halves per-step matrix bytes)
    k_cvt<<<(int)(PELEMS / (256 * 8)), 256, 0, stream>>>(proj, proj16);

  // init: x = best_x = x_start; best_obj0 = qp_obj(x_start)
  k_init<<<128, 256, 0, stream>>>(x_start, x, best_x, flag);
  k_quad<<<dim3(64, BB), 256, 0, stream>>>(src, dst, ew, x, quadpart);
  k_lin_init<<<128, 256, 0, stream>>>(q, x, linpart);
  k_objfin<<<1, 32, 0, stream>>>(quadpart, linpart, best_obj, flag, obj_sol,
                                 out_best, out_gaps, -1);

  float tau = 1.0f;   // INIT_TAU, mirrored on host (pure function of step)
  for (int s = 0; s < STEPS; ++s) {
    k_dir<<<128, 256, 0, stream>>>(x, rhs, flag, best_x, d, tau);
    tau = fmaxf(tau * 0.5f, 1e-5f);
    if (use16) {
      k_coef_t<__bf16><<<dim3(FF / 16, BB), 256, 0, stream>>>(proj16, d, coef);
      k_pred_t<__bf16><<<dim3(NPG / 16, BB), 256, 0, stream>>>(proj16, coef, x, pred, minpart);
    } else {
      k_coef_t<float><<<dim3(FF / 16, BB), 256, 0, stream>>>(proj, d, coef);
      k_pred_t<float><<<dim3(NPG / 16, BB), 256, 0, stream>>>(proj, coef, x, pred, minpart);
    }
    k_alpha<<<BB, 128, 0, stream>>>(minpart, alpha);
    k_xupd<<<128, 256, 0, stream>>>(x, pred, alpha, q, linpart);
    k_quad<<<dim3(64, BB), 256, 0, stream>>>(src, dst, ew, x, quadpart);
    k_objfin<<<1, 32, 0, stream>>>(quadpart, linpart, best_obj, flag, obj_sol,
                                   out_best, out_gaps, s);
  }
  k_flush<<<128, 256, 0, stream>>>(x, flag, best_x);
}
